// ContrastiveLoss_66666482368608
// MI455X (gfx1250) — matrix operand order repaired and checked by reading, not compiled, not run on hardware
//
#include <hip/hip_runtime.h>

// ---------------------------------------------------------------------------
// Contrastive loss on MI455X (gfx1250):
//   flat = features.reshape(6144, 1024); N = flat / max(||row||, 1e-8)
//   loss = mean over strict-upper-triangle (r<c) of
//          0.5*( y*d^2 + (1-y)*relu(0.5-d)^2 ),  d = 1 - (N N^T)[r,c]
// GEMM on v_wmma_f32_16x16x32_f16, async global->LDS double-buffered staging,
// loss fused into the epilogue. Deterministic (no atomics).
// ---------------------------------------------------------------------------

typedef _Float16 v16h __attribute__((ext_vector_type(16)));
typedef _Float16 v8h  __attribute__((ext_vector_type(8)));
typedef float    v8f  __attribute__((ext_vector_type(8)));

#define DDIM 1024          // feature dim
#define RDIM 6144          // 3*B rows
#define BT   128           // macro tile (rows and cols)
#define KT   32            // WMMA K
#define KS   64            // staged K panel (2 WMMA sub-steps per barrier)
#define NBLK (RDIM / BT)   // 48
#define NTILES (NBLK * (NBLK + 1) / 2)       // 1176 upper-tri macro tiles
#define LSTR 72            // LDS row stride in halfs (144B: 16B-aligned, bank-friendly)
#define PAIRS 18871296.0   // 6144*6143/2

// ---------------------------------------------------------------------------
// Kernel 1: per-row L2 normalize f32 -> f16, and build label vector L[6144].
// ---------------------------------------------------------------------------
__global__ __launch_bounds__(256) void cl_normalize(
    const float* __restrict__ feat, const int* __restrict__ labels,
    const int* __restrict__ neg_labels, _Float16* __restrict__ Nmat,
    int* __restrict__ L) {
  const int r = blockIdx.x;
  const float* row = feat + (size_t)r * DDIM;
  float s = 0.f;
  for (int k = threadIdx.x; k < DDIM; k += 256) {
    float v = row[k];
    s += v * v;
  }
  __shared__ float red[256];
  red[threadIdx.x] = s;
  __syncthreads();
  for (int off = 128; off > 0; off >>= 1) {
    if (threadIdx.x < off) red[threadIdx.x] += red[threadIdx.x + off];
    __syncthreads();
  }
  const float inv = 1.f / fmaxf(sqrtf(red[0]), 1e-8f);
  for (int k = threadIdx.x; k < DDIM; k += 256)
    Nmat[(size_t)r * DDIM + k] = (_Float16)(row[k] * inv);
  if (threadIdx.x == 0) {
    int item = r / 3, sub = r - 3 * item;
    L[r] = (sub < 2) ? labels[item] : neg_labels[item];
  }
}

// ---------------------------------------------------------------------------
// Kernel 2: fused gram + loss over one 128x128 macro tile of the upper
// triangle. 8 waves; each wave owns a 32x64 patch = 2x4 WMMA accumulators.
// Double-buffered async global->LDS staging (ASYNCcnt).
// ---------------------------------------------------------------------------
__global__ __launch_bounds__(256) void cl_gram_loss(
    const _Float16* __restrict__ Nmat, const int* __restrict__ L,
    float* __restrict__ partial) {
  // linear block id -> (bi, bj) with bi <= bj over the 48x48 block grid
  int t = blockIdx.x;
  int bi = 0;
  while (t >= NBLK - bi) { t -= NBLK - bi; ++bi; }
  const int bj = bi + t;
  const int rowA0 = bi * BT;
  const int rowB0 = bj * BT;

  __shared__ _Float16 As[2][BT][LSTR];
  __shared__ _Float16 Bs[2][BT][LSTR];
  __shared__ int Lrow[BT];
  __shared__ int Lcol[BT];

  const int tid  = threadIdx.x;
  const int lane = tid & 31;
  const int wid  = tid >> 5;   // 0..7
  const int wm   = wid & 3;    // 4 row groups of 32
  const int wn   = wid >> 2;   // 2 col groups of 64
  const int lsel = lane >> 4;  // half-wave select (ISA fragment layouts)
  const int lr   = lane & 15;

  // cache this tile's labels in LDS (synced by the first loop barrier)
  if (tid < BT) Lrow[tid] = L[rowA0 + tid];
  else          Lcol[tid - BT] = L[rowB0 + (tid - BT)];

  v8f acc[2][4];
#pragma unroll
  for (int am = 0; am < 2; ++am)
#pragma unroll
    for (int bn = 0; bn < 4; ++bn) acc[am][bn] = (v8f)0.f;

  // staging assignment: thread -> (row tid>>1, 32-half chunk (tid&1)*32),
  // 4 async b128 per matrix per thread = full 128x64 panel per stage.
  const int sr = tid >> 1;
  const int sc = (tid & 1) * 32;

  auto stage = [&](int buf, int k0) {
    const _Float16* ga = Nmat + (size_t)(rowA0 + sr) * DDIM + k0 + sc;
    const _Float16* gb = Nmat + (size_t)(rowB0 + sr) * DDIM + k0 + sc;
    unsigned la = (unsigned)(__SIZE_TYPE__)&As[buf][sr][sc];
    unsigned lb = (unsigned)(__SIZE_TYPE__)&Bs[buf][sr][sc];
#pragma unroll
    for (int c = 0; c < 32; c += 8) {
      asm volatile("global_load_async_to_lds_b128 %0, %1, off"
                   :: "v"(la + c * 2), "v"(ga + c) : "memory");
      asm volatile("global_load_async_to_lds_b128 %0, %1, off"
                   :: "v"(lb + c * 2), "v"(gb + c) : "memory");
    }
  };

  stage(0, 0);
  int cur = 0;
  for (int k0 = 0; k0 < DDIM; k0 += KS) {
    asm volatile("s_wait_asynccnt 0x0" ::: "memory");  // my panel landed in LDS
    __syncthreads();                                   // everyone's panel landed
    if (k0 + KS < DDIM) stage(cur ^ 1, k0 + KS);       // prefetch next panel

#pragma unroll
    for (int ks = 0; ks < KS; ks += KT) {
      // A fragment per-lane layout (16-bit A 16x32, ISA 7.12.2):
      //   lanes 0-15: K 0-7 then 16-23 ; lanes 16-31: K 8-15 then 24-31
      v16h afrag[2];
#pragma unroll
      for (int am = 0; am < 2; ++am) {
        const int arow = wm * 32 + am * 16 + lr;
        v16h a;
        *(v8h*)&a       = *(const v8h*)&As[cur][arow][ks + lsel * 8];
        *((v8h*)&a + 1) = *(const v8h*)&As[cur][arow][ks + 16 + lsel * 8];
        afrag[am] = a;
      }
      // B fragment per-lane layout (16-bit B 32x16), same K order as A:
      //   lanes 0-15: K 0-7 then 16-23 ; lanes 16-31: K 8-15 then 24-31
#pragma unroll
      for (int bn = 0; bn < 4; ++bn) {
        const int bcol = wn * 64 + bn * 16 + lr;
        v16h b;
        *(v8h*)&b       = *(const v8h*)&Bs[cur][bcol][ks + lsel * 8];
        *((v8h*)&b + 1) = *(const v8h*)&Bs[cur][bcol][ks + 16 + lsel * 8];
#pragma unroll
        for (int am = 0; am < 2; ++am) {
          acc[am][bn] = __builtin_amdgcn_wmma_f32_16x16x32_f16(
              false, afrag[am], false, b, (short)0, acc[am][bn], false, false);
        }
      }
    }
    __syncthreads();  // all waves done reading buf 'cur' before it is restaged
    cur ^= 1;
  }

  // Epilogue: C element (VGPR v, lane) -> M = v + 8*lsel, N = lane&15.
  float lsum = 0.f;
#pragma unroll
  for (int am = 0; am < 2; ++am) {
#pragma unroll
    for (int bn = 0; bn < 4; ++bn) {
      const int cl = wn * 64 + bn * 16 + lr;         // local col
      const int rl = wm * 32 + am * 16 + lsel * 8;   // local row base
      const int c = rowB0 + cl;
      const int lc = Lcol[cl];
#pragma unroll
      for (int v = 0; v < 8; ++v) {
        const int r = rowA0 + rl + v;
        if (r < c) {  // strict upper triangle only
          const float d = 1.f - acc[am][bn][v];
          const float m = fmaxf(0.5f - d, 0.f);
          lsum += (Lrow[rl + v] == lc) ? 0.5f * d * d : 0.5f * m * m;
        }
      }
    }
  }

  __shared__ float red[256];
  red[tid] = lsum;
  __syncthreads();
  for (int off = 128; off > 0; off >>= 1) {
    if (tid < off) red[tid] += red[tid + off];
    __syncthreads();
  }
  if (tid == 0) partial[blockIdx.x] = red[0];
}

// ---------------------------------------------------------------------------
// Kernel 3: deterministic final reduction of block partials in double.
// ---------------------------------------------------------------------------
__global__ __launch_bounds__(256) void cl_reduce(const float* __restrict__ partial,
                                                 int n, float* __restrict__ out) {
  __shared__ double red[256];
  double s = 0.0;
  for (int i = threadIdx.x; i < n; i += 256) s += (double)partial[i];
  red[threadIdx.x] = s;
  __syncthreads();
  for (int off = 128; off > 0; off >>= 1) {
    if (threadIdx.x < off) red[threadIdx.x] += red[threadIdx.x + off];
    __syncthreads();
  }
  if (threadIdx.x == 0) out[0] = (float)(red[0] / PAIRS);
}

extern "C" void kernel_launch(void* const* d_in, const int* in_sizes, int n_in,
                              void* d_out, int out_size, void* d_ws, size_t ws_size,
                              hipStream_t stream) {
  (void)in_sizes; (void)n_in; (void)out_size; (void)ws_size;
  const float* feat       = (const float*)d_in[0];  // [2048,3,1024] f32
  const int*   labels     = (const int*)d_in[1];    // [2048] i32
  const int*   neg_labels = (const int*)d_in[2];    // [2048] i32
  float*       out        = (float*)d_out;          // scalar f32

  // Workspace layout (all 16B aligned):
  //   [0)                 N     : 6144*1024 f16  = 12,582,912 B
  //   [12582912)          L     : 6144 i32       =     24,576 B
  //   [12607488)          part  : 1176 f32       =      4,704 B
  char* ws = (char*)d_ws;
  _Float16* Nmat = (_Float16*)(ws);
  int*      L    = (int*)(ws + (size_t)RDIM * DDIM * 2);
  float*    part = (float*)(ws + (size_t)RDIM * DDIM * 2 + (size_t)RDIM * 4);

  cl_normalize<<<RDIM, 256, 0, stream>>>(feat, labels, neg_labels, Nmat, L);
  cl_gram_loss<<<NTILES, 256, 0, stream>>>(Nmat, L, part);
  cl_reduce<<<1, 256, 0, stream>>>(part, NTILES, out);
}
